// PPGN_23888608100580
// MI455X (gfx1250) — matrix operand order repaired and checked by reading, not compiled, not run on hardware
//
#include <hip/hip_runtime.h>
#include <hip/hip_bf16.h>

// ---------------------------------------------------------------------------
// PPGN forward for MI455X (gfx1250, wave32, WMMA).
// B=64, N=128, F_IN=16, EMB=64, NCLS=10.  M = B*N*N = 1,048,576 pixels.
// All conv1x1 + NxN matmuls run on v_wmma_f32_16x16x32_bf16.
// ---------------------------------------------------------------------------

typedef __attribute__((ext_vector_type(16))) __bf16    v16bf;
typedef __attribute__((ext_vector_type(8)))  float     v8f;
typedef __attribute__((ext_vector_type(4)))  unsigned  v4u;

static constexpr int    Bsz  = 64;
static constexpr int    Nn   = 128;
static constexpr int    FIN  = 16;
static constexpr int    EMB  = 64;
static constexpr long   MPIX = (long)Bsz * Nn * Nn;      // 1,048,576
static constexpr int    EDG  = 131072;                   // B * N * DEG

__device__ __forceinline__ __bf16 f2bf(float f) { return (__bf16)f; }

union Frag { v4u u[2]; v16bf v; };

// A-fragment for V_WMMA_F32_16X16X32_BF16 from a row-major [*, K] matrix.
// Lane L: m = L%16 (row), h = L/16. Element j -> k=h*16+j.
// 'row' points at this lane's row, k-offset already applied. Needs 16B alignment.
__device__ __forceinline__ v16bf afrag_load(const __bf16* row, int lane) {
  int h = lane >> 4;
  const v4u* p = (const v4u*)row;
  Frag f; f.u[0] = p[2 * h]; f.u[1] = p[2 * h + 1];
  return f.v;
}

// 16 contiguous bf16 (used for pre-packed B fragments & LDS-transposed B tiles).
__device__ __forceinline__ v16bf frag_contig(const __bf16* p) {
  const v4u* q = (const v4u*)p;
  Frag f; f.u[0] = q[0]; f.u[1] = q[1];
  return f.v;
}

__device__ __forceinline__ v8f wmma_bf16(v16bf a, v16bf b, v8f c) {
  // (neg_a, A, neg_b, B, c_mod, C, reuse_a, reuse_b)
  return __builtin_amdgcn_wmma_f32_16x16x32_bf16(false, a, false, b, (short)0, c,
                                                 false, false);
}

// ---------------------------------------------------------------------------
// Setup kernels
// ---------------------------------------------------------------------------
__global__ void k_zero_f32(float* p, long n) {
  long i = (long)blockIdx.x * blockDim.x + threadIdx.x;
  if (i < n) p[i] = 0.f;
}

// dense adjacency: adj[b][i][j] += 1 for each edge (graphs are all size N).
__global__ void k_scatter(const int* __restrict__ ei, float* __restrict__ adj) {
  int e = blockIdx.x * blockDim.x + threadIdx.x;
  if (e >= EDG) return;
  int s = ei[e], d = ei[EDG + e];
  int g = s >> 7, i = s & 127, j = d & 127;
  atomicAdd(&adj[((long)g * Nn + i) * Nn + j], 1.0f);
}

// z0[b][u][v][0]   = adj[b][v][u]                (transpose(z,1,3) in reference)
// z0[b][u][v][1+f] = (u==v) ? x[b][v][f] : 0 ;  channels padded 17 -> 32, bf16.
__global__ void k_build_z0(const float* __restrict__ adj, const float* __restrict__ x,
                           __bf16* __restrict__ z0) {
  long p = (long)blockIdx.x * blockDim.x + threadIdx.x;
  if (p >= MPIX) return;
  int b = (int)(p >> 14), u = (int)((p >> 7) & 127), v = (int)(p & 127);
  alignas(16) __bf16 row[32];
  row[0] = f2bf(adj[((long)b * Nn + v) * Nn + u]);
#pragma unroll
  for (int f = 0; f < FIN; ++f)
    row[1 + f] = (u == v) ? f2bf(x[((long)b * Nn + v) * FIN + f]) : f2bf(0.f);
#pragma unroll
  for (int f = 1 + FIN; f < 32; ++f) row[f] = f2bf(0.f);
  v4u* dst = (v4u*)(z0 + p * 32);
  const v4u* s = (const v4u*)row;
  dst[0] = s[0]; dst[1] = s[1]; dst[2] = s[2]; dst[3] = s[3];
}

// Pack fp32 weight [Cout x CinTotal] slice [cbeg, cbeg+Kpad) (zeros past cend)
// into WMMA B-fragment lane order, bf16:  idx = ((kt*nq+q)*32+lane)*16 + j,
// value = W[o = q*16+lane%16][c = cbeg + kt*32 + (lane/16)*16 + j].
__global__ void k_pack_w(const float* __restrict__ W, __bf16* __restrict__ out,
                         int Cout, int CinTotal, int cbeg, int cend, int Kpad) {
  int nq = Cout / 16;
  int total = (Kpad / 32) * nq * 32 * 16;
  for (int idx = threadIdx.x; idx < total; idx += blockDim.x) {
    int j    = idx & 15;
    int lane = (idx >> 4) & 31;
    int q    = (idx >> 9) % nq;
    int kt   = (idx >> 9) / nq;
    int o = q * 16 + (lane & 15);
    int c = cbeg + kt * 32 + (lane >> 4) * 16 + j;
    float v = (c < cend) ? W[o * CinTotal + c] : 0.f;
    out[idx] = f2bf(v);
  }
}

// ---------------------------------------------------------------------------
// Fused 2-layer MLP: Y = relu(W2 * relu(W1 * X + b1) + b2).
// X channels-last bf16 [M, K1]; intermediate (128 x 64 bf16) lives in LDS only.
// Output plane-major bf16 [b][o][N*N] (feeds the batched matmul).
// Block: 256 threads = 8 waves, 128 pixels; wave owns 16 rows, full Cout=64.
// ---------------------------------------------------------------------------
template <int K1>
__global__ void k_mlp2(const __bf16* __restrict__ X,
                       const __bf16* __restrict__ pack1, const float* __restrict__ b1,
                       const __bf16* __restrict__ pack2, const float* __restrict__ b2,
                       __bf16* __restrict__ out) {
  __shared__ __bf16 lds[128 * 64];
  int t = threadIdx.x, lane = t & 31, wave = t >> 5;
  int h = lane >> 4, nl = lane & 15;
  long m0 = (long)blockIdx.x * 128 + wave * 16;

  // ---- stage 1: K1 -> 64, relu, park in LDS (wave-private rows) ----
  v8f acc[4] = {};
  const __bf16* arow = X + (m0 + nl) * K1;
#pragma unroll
  for (int kt = 0; kt < K1 / 32; ++kt) {
    v16bf a = afrag_load(arow + kt * 32, lane);
#pragma unroll
    for (int q = 0; q < 4; ++q) {
      v16bf b = frag_contig(pack1 + (((long)(kt * 4 + q) * 32 + lane) << 4));
      acc[q] = wmma_bf16(a, b, acc[q]);
    }
  }
#pragma unroll
  for (int q = 0; q < 4; ++q) {
    float bias = b1[q * 16 + nl];
#pragma unroll
    for (int r = 0; r < 8; ++r) {
      int m = r + 8 * h;  // D layout: lane L holds (m = r + 8*(L/16), n = L%16)
      lds[(wave * 16 + m) * 64 + q * 16 + nl] = f2bf(fmaxf(acc[q][r] + bias, 0.f));
    }
  }
  // wave-private rows: per-wave LDS ordering (DScnt in-order) suffices, no barrier.

  // ---- stage 2: 64 -> 64, relu, store plane-major ----
  v8f acc2[4] = {};
  const __bf16* lrow = &lds[(wave * 16 + nl) * 64];
#pragma unroll
  for (int kt = 0; kt < 2; ++kt) {
    v16bf a = afrag_load(lrow + kt * 32, lane);
#pragma unroll
    for (int q = 0; q < 4; ++q) {
      v16bf b = frag_contig(pack2 + (((long)(kt * 4 + q) * 32 + lane) << 4));
      acc2[q] = wmma_bf16(a, b, acc2[q]);
    }
  }
#pragma unroll
  for (int q = 0; q < 4; ++q) {
    float bias = b2[q * 16 + nl];
#pragma unroll
    for (int r = 0; r < 8; ++r) {
      int m = r + 8 * h;
      long P = m0 + m;
      long img = P >> 14, pix = P & 16383;
      out[(img * 64 + (q * 16 + nl)) * 16384 + pix] =
          f2bf(fmaxf(acc2[q][r] + bias, 0.f));
    }
  }
}

// ---------------------------------------------------------------------------
// Skip conv: Y = W0*A0 + W1*A1 + bias (no relu), concat handled as 2 K-loops.
// A0 cl [M,K0], A1 cl [M,64].  Output channels-last bf16 [M,64].
// ---------------------------------------------------------------------------
template <int K0>
__global__ void k_skip(const __bf16* __restrict__ A0, const __bf16* __restrict__ pack0,
                       const __bf16* __restrict__ A1, const __bf16* __restrict__ pack1,
                       const float* __restrict__ bias, __bf16* __restrict__ out) {
  int t = threadIdx.x, lane = t & 31, wave = t >> 5;
  int h = lane >> 4, nl = lane & 15;
  long m0 = (long)blockIdx.x * 128 + wave * 16;
  v8f acc[4] = {};
  const __bf16* r0 = A0 + (m0 + nl) * K0;
#pragma unroll
  for (int kt = 0; kt < K0 / 32; ++kt) {
    v16bf a = afrag_load(r0 + kt * 32, lane);
#pragma unroll
    for (int q = 0; q < 4; ++q) {
      v16bf b = frag_contig(pack0 + (((long)(kt * 4 + q) * 32 + lane) << 4));
      acc[q] = wmma_bf16(a, b, acc[q]);
    }
  }
  const __bf16* r1 = A1 + (m0 + nl) * 64;
#pragma unroll
  for (int kt = 0; kt < 2; ++kt) {
    v16bf a = afrag_load(r1 + kt * 32, lane);
#pragma unroll
    for (int q = 0; q < 4; ++q) {
      v16bf b = frag_contig(pack1 + (((long)(kt * 4 + q) * 32 + lane) << 4));
      acc[q] = wmma_bf16(a, b, acc[q]);
    }
  }
#pragma unroll
  for (int q = 0; q < 4; ++q) {
    float bv = bias[q * 16 + nl];
#pragma unroll
    for (int r = 0; r < 8; ++r) {
      int m = r + 8 * h;
      out[(m0 + m) * 64 + q * 16 + nl] = f2bf(acc[q][r] + bv);
    }
  }
}

// ---------------------------------------------------------------------------
// Batched 128x128x128 matmul: one block per (b,c) plane. A,B plane-major bf16.
// B tile (32x128) is transposed through LDS so B fragments are contiguous.
// Output channels-last bf16 mult[b][u][v][c] (feeds the skip conv).
// ---------------------------------------------------------------------------
__global__ void k_bmm(const __bf16* __restrict__ Ap, const __bf16* __restrict__ Bp,
                      __bf16* __restrict__ outCl) {
  __shared__ __bf16 bT[128 * 32];  // [col][krel]  8 KB
  int plane = blockIdx.x;          // b*64 + c
  long pofs = (long)plane * 16384;
  const __bf16* Am = Ap + pofs;
  const __bf16* Bm = Bp + pofs;
  int t = threadIdx.x, lane = t & 31, wave = t >> 5;
  int h = lane >> 4, nl = lane & 15;
  int b = plane >> 6, c = plane & 63;

  v8f acc[8] = {};
  for (int kt = 0; kt < 4; ++kt) {
    // stage B rows [kt*32, kt*32+32) transposed: coalesced global -> ds stores
#pragma unroll
    for (int i = 0; i < 8; ++i) {
      int idx2 = i * 256 + t;          // 2048 col-pairs
      int r = idx2 >> 6, cp = (idx2 & 63) * 2;
      const __bf16* src = Bm + (kt * 32 + r) * 128 + cp;
      bT[(cp + 0) * 32 + r] = src[0];
      bT[(cp + 1) * 32 + r] = src[1];
    }
    __syncthreads();
    const __bf16* arow = Am + (wave * 16 + nl) * 128 + kt * 32;
    v16bf a = afrag_load(arow, lane);
#pragma unroll
    for (int q = 0; q < 8; ++q) {
      v16bf bf = frag_contig(&bT[(q * 16 + nl) * 32 + h * 16]);
      acc[q] = wmma_bf16(a, bf, acc[q]);
    }
    __syncthreads();
  }
#pragma unroll
  for (int q = 0; q < 8; ++q)
#pragma unroll
    for (int r = 0; r < 8; ++r) {
      int u = wave * 16 + r + 8 * h, v = q * 16 + nl;
      outCl[(((long)b * 128 + u) * 128 + v) * 64 + c] = f2bf(acc[q][r]);
    }
}

// ---------------------------------------------------------------------------
// diag/offdiag mean-pool: h[b][0:64]=mean_diag, h[b][64:128]=mean_off (fp32)
// ---------------------------------------------------------------------------
__global__ void k_pool(const __bf16* __restrict__ z, float* __restrict__ hbuf) {
  int b = blockIdx.x, t = threadIdx.x;
  int c = t & 63, s = t >> 6;  // 4 pixel stripes per channel
  const __bf16* base = z + (long)b * 16384 * 64;
  float sum = 0.f, dsum = 0.f;
  for (int p = s * 4096; p < (s + 1) * 4096; ++p) {
    float v = (float)base[(long)p * 64 + c];
    sum += v;
    if ((p >> 7) == (p & 127)) dsum += v;
  }
  __shared__ float ls[256], ld[256];
  ls[t] = sum; ld[t] = dsum;
  __syncthreads();
  if (t < 64) {
    float S = ls[t] + ls[t + 64] + ls[t + 128] + ls[t + 192];
    float D = ld[t] + ld[t + 64] + ld[t + 128] + ld[t + 192];
    float md = D / 128.f;
    hbuf[b * 128 + t]      = md;
    hbuf[b * 128 + 64 + t] = (S - md * 128.f) / (128.f * 127.f);
  }
}

// fc0(relu) -> fc1 -> log_softmax.  64 blocks x 64 threads (tiny).
__global__ void k_head(const float* __restrict__ hbuf,
                       const float* __restrict__ w0, const float* __restrict__ b0,
                       const float* __restrict__ w1, const float* __restrict__ b1,
                       float* __restrict__ out) {
  int b = blockIdx.x, t = threadIdx.x;
  __shared__ float hv[128], s1[64], lg[10], red[2];
  hv[t] = hbuf[b * 128 + t];
  hv[64 + t] = hbuf[b * 128 + 64 + t];
  __syncthreads();
  float a = b0[t];
  for (int k = 0; k < 128; ++k) a += w0[t * 128 + k] * hv[k];
  s1[t] = fmaxf(a, 0.f);
  __syncthreads();
  if (t < 10) {
    float a2 = b1[t];
    for (int k = 0; k < 64; ++k) a2 += w1[t * 64 + k] * s1[k];
    lg[t] = a2;
  }
  __syncthreads();
  if (t == 0) {
    float m = lg[0];
    for (int i = 1; i < 10; ++i) m = fmaxf(m, lg[i]);
    float se = 0.f;
    for (int i = 0; i < 10; ++i) se += expf(lg[i] - m);
    red[0] = m; red[1] = logf(se);
  }
  __syncthreads();
  if (t < 10) out[b * 10 + t] = lg[t] - red[0] - red[1];
}

// ---------------------------------------------------------------------------
// Workspace layout (bytes)
// ---------------------------------------------------------------------------
static constexpr size_t OFF_ADJ  = 0;                            // 4 MiB f32
static constexpr size_t OFF_PACK = (size_t)4 << 20;              // 12 * 8 KiB
static constexpr size_t OFF_H    = OFF_PACK + 12 * 8192;         // 32 KiB f32
static constexpr size_t OFF_Z0   = (size_t)5 << 20;              // 64 MiB bf16 [M,32]
static constexpr size_t OFF_P1   = OFF_Z0 + (size_t)MPIX * 32 * 2;
static constexpr size_t OFF_P2   = OFF_P1 + (size_t)MPIX * 64 * 2;
static constexpr size_t OFF_P3   = OFF_P2 + (size_t)MPIX * 64 * 2;
static constexpr size_t OFF_P4   = OFF_P3 + (size_t)MPIX * 64 * 2;

extern "C" void kernel_launch(void* const* d_in, const int* in_sizes, int n_in,
                              void* d_out, int out_size, void* d_ws, size_t ws_size,
                              hipStream_t stream) {
  // jax pytree flatten order (dict keys sorted): x, edge_index, batch,
  // fc0(w,b), fc1(w,b), rb0_m1[0](w,b), rb0_m1[1](w,b), rb0_m2[0..1],
  // rb0_skip, rb1_m1[0..1], rb1_m2[0..1], rb1_skip
  const float* x    = (const float*)d_in[0];
  const int*   ei   = (const int*)d_in[1];
  const float* fc0w = (const float*)d_in[3];  const float* fc0b = (const float*)d_in[4];
  const float* fc1w = (const float*)d_in[5];  const float* fc1b = (const float*)d_in[6];
  const float* w[8]; const float* bia[8];
  for (int i = 0; i < 8; ++i) { w[i] = (const float*)d_in[7 + 2*i]; bia[i] = (const float*)d_in[8 + 2*i]; }
  // w[0..1]=rb0_m1, w[2..3]=rb0_m2, w[4]=rb0_skip, w[5..6]=rb1_m1, w[7]=rb1_m2[0]
  const float* rb1m2_1w = (const float*)d_in[23]; const float* rb1m2_1b = (const float*)d_in[24];
  const float* rb1skw   = (const float*)d_in[25]; const float* rb1skb   = (const float*)d_in[26];

  char* ws = (char*)d_ws;
  float*  adj = (float*)(ws + OFF_ADJ);
  float*  hbf = (float*)(ws + OFF_H);
  __bf16* z0  = (__bf16*)(ws + OFF_Z0);
  __bf16* P1  = (__bf16*)(ws + OFF_P1);
  __bf16* P2  = (__bf16*)(ws + OFF_P2);
  __bf16* P3  = (__bf16*)(ws + OFF_P3);
  __bf16* P4  = (__bf16*)(ws + OFF_P4);
  auto pk = [&](int i) { return (__bf16*)(ws + OFF_PACK + (size_t)i * 8192); };

  // --- setup ---
  k_zero_f32<<<4096, 256, 0, stream>>>(adj, MPIX);
  k_scatter<<<EDG / 256, 256, 0, stream>>>(ei, adj);
  k_build_z0<<<4096, 256, 0, stream>>>(adj, x, z0);

  // --- pack weights into B-fragment order ---
  k_pack_w<<<1, 256, 0, stream>>>(w[0], pk(0), 64, 17, 0, 17, 32);   // rb0_m1 L0
  k_pack_w<<<1, 256, 0, stream>>>(w[1], pk(1), 64, 64, 0, 64, 64);   // rb0_m1 L1
  k_pack_w<<<1, 256, 0, stream>>>(w[2], pk(2), 64, 17, 0, 17, 32);   // rb0_m2 L0
  k_pack_w<<<1, 256, 0, stream>>>(w[3], pk(3), 64, 64, 0, 64, 64);   // rb0_m2 L1
  k_pack_w<<<1, 256, 0, stream>>>(w[4], pk(4), 64, 81, 0, 17, 32);   // rb0_skip (z part)
  k_pack_w<<<1, 256, 0, stream>>>(w[4], pk(5), 64, 81, 17, 81, 64);  // rb0_skip (mult part)
  k_pack_w<<<1, 256, 0, stream>>>(w[5], pk(6), 64, 64, 0, 64, 64);   // rb1_m1 L0
  k_pack_w<<<1, 256, 0, stream>>>(w[6], pk(7), 64, 64, 0, 64, 64);   // rb1_m1 L1
  k_pack_w<<<1, 256, 0, stream>>>(w[7], pk(8), 64, 64, 0, 64, 64);   // rb1_m2 L0
  k_pack_w<<<1, 256, 0, stream>>>(rb1m2_1w, pk(9), 64, 64, 0, 64, 64); // rb1_m2 L1
  k_pack_w<<<1, 256, 0, stream>>>(rb1skw, pk(10), 64, 128, 0, 64, 64);  // rb1_skip (z)
  k_pack_w<<<1, 256, 0, stream>>>(rb1skw, pk(11), 64, 128, 64, 128, 64);// rb1_skip (mult)

  const int GEMM_GRID = (int)(MPIX / 128);  // 8192

  // --- regular block 0 ---
  k_mlp2<32><<<GEMM_GRID, 256, 0, stream>>>(z0, pk(0), bia[0], pk(1), bia[1], P1); // m1 plane
  k_mlp2<32><<<GEMM_GRID, 256, 0, stream>>>(z0, pk(2), bia[2], pk(3), bia[3], P2); // m2 plane
  k_bmm<<<Bsz * EMB, 256, 0, stream>>>(P1, P2, P3);                                // mult cl
  k_skip<32><<<GEMM_GRID, 256, 0, stream>>>(z0, pk(4), P3, pk(5), bia[4], P4);     // z1 cl

  // --- regular block 1 ---
  k_mlp2<64><<<GEMM_GRID, 256, 0, stream>>>(P4, pk(6), bia[5], pk(7), bia[6], P1);
  k_mlp2<64><<<GEMM_GRID, 256, 0, stream>>>(P4, pk(8), bia[7], pk(9), rb1m2_1b, P2);
  k_bmm<<<Bsz * EMB, 256, 0, stream>>>(P1, P2, P3);
  k_skip<64><<<GEMM_GRID, 256, 0, stream>>>(P4, pk(10), P3, pk(11), rb1skb, P1);   // z2 cl

  // --- pool + head ---
  k_pool<<<Bsz, 256, 0, stream>>>(P1, hbf);
  k_head<<<Bsz, 64, 0, stream>>>(hbf, fc0w, fc0b, fc1w, fc1b, (float*)d_out);
  (void)in_sizes; (void)n_in; (void)out_size; (void)ws_size;
}
